// GaussianRenderHead_79783312490969
// MI455X (gfx1250) — matrix operand order repaired and checked by reading, not compiled, not run on hardware
//
#include <hip/hip_runtime.h>
#include <hip/hip_bf16.h>

typedef __attribute__((ext_vector_type(16))) _Float16 v16h;
typedef __attribute__((ext_vector_type(8)))  float    v8f;
typedef unsigned int u32;

#define H_IMG 480
#define W_IMG 640
#define HW    (H_IMG * W_IMG)
#define NG    256
#define CF    256
#define FARV  80.0f

// ---------------- workspace layout (bytes) ----------------
// sorted per-gaussian params: 7 arrays of 256 f32: [0]=0.5A [1]=0.5C [2]=B [3]=u [4]=v [5]=op [6]=tz
#define OFF_PAR    0
#define OFF_FEATT  8192      // f16 [256 channel][256 gauss]  (gauss axis depth-sorted)
#define OFF_W1T    139264    // f16 [128 n][256 k]
#define OFF_W2T    204800    // f16 [128 n][128 k]
#define OFF_W3T    237568    // f16 [16 n][128 k], rows n>=12 are zero

// ---------------- LDS layout (bytes) ----------------
// f16 row strides padded so (rowDwords % 64) != 0  -> avoids 16/32-way bank conflicts,
// while staying 16B-aligned for B128 staging stores.
#define FSTR   264           // f16 stride for 256-wide rows (132 dwords == 4 mod 64)
#define FSTRD  132
#define HSTR   136           // f16 stride for 128-wide rows (68 dwords == 4 mod 64)
#define HSTRD  68
#define L_WGT    0                       // f16 [32 px][FSTR]
#define L_REND   16896                   // f16 [32 px][FSTR]
#define L_H1     33792                   // f16 [32 px][HSTR]
#define L_H2     42496                   // f16 [32 px][HSTR]
#define L_SCAN   51200                   // f32 [32 px][9]
#define L_DEPTH  52352                   // f32 [32 px]
#define L_FEAT   52480                   // f16 [256 ch][FSTR]
#define SMEM_BYTES (52480 + 256 * FSTR * 2)   // 187648 <= 320KB/WGP

union Frag16 { v16h h; u32 u[8]; };

// A-matrix 16x32 f16 fragment: lane L: g=L/16, M=L%16; VGPR v: K = 2v + 8g (+8 if v>=4)
__device__ __forceinline__ v16h load_afrag(const u32* base, int rowDwords, int row, int kk, int g) {
  Frag16 f;
#pragma unroll
  for (int v = 0; v < 8; ++v) {
    int koff = 2 * v + 8 * g + ((v >= 4) ? 8 : 0);
    f.u[v] = base[row * rowDwords + ((kk * 32 + koff) >> 1)];
  }
  return f.h;
}

// B-matrix 32x16 f16 fragment: lane L: N=L%16, g=L/16; VGPR v: K = 2v + 8g (+8 if v>=4)
__device__ __forceinline__ v16h load_bfrag(const u32* base, int rowDwords, int n, int kk, int g) {
  Frag16 f;
#pragma unroll
  for (int v = 0; v < 8; ++v) {
    int k = kk * 32 + 2 * v + 8 * g + ((v >= 4) ? 8 : 0);
    f.u[v] = base[n * rowDwords + (k >> 1)];
  }
  return f.h;
}

// ================= prep kernel 1: per-gaussian projection + depth sort =================
__global__ void prep_gauss(const float* __restrict__ means3d, const float* __restrict__ features,
                           const float* __restrict__ opac,    const float* __restrict__ scales,
                           const float* __restrict__ rot,     const float* __restrict__ camK,
                           const float* __restrict__ pose,    char* __restrict__ ws) {
  __shared__ float keys[NG];
  int i = threadIdx.x;

  float fx = camK[0], cx = camK[2], fy = camK[4], cy = camK[5];
  float Rw[3][3], tw[3];
#pragma unroll
  for (int r = 0; r < 3; ++r) {
#pragma unroll
    for (int c = 0; c < 3; ++c) Rw[r][c] = pose[r * 4 + c];
    tw[r] = pose[r * 4 + 3];
  }

  float mx = means3d[i * 3 + 0], my = means3d[i * 3 + 1], mz = means3d[i * 3 + 2];
  float tcx = Rw[0][0] * mx + Rw[0][1] * my + Rw[0][2] * mz + tw[0];
  float tcy = Rw[1][0] * mx + Rw[1][1] * my + Rw[1][2] * mz + tw[1];
  float tcz = Rw[2][0] * mx + Rw[2][1] * my + Rw[2][2] * mz + tw[2];
  bool valid = tcz > 0.01f;
  float tzs = valid ? tcz : 1.0f;

  float qw = rot[i * 4 + 0], qx = rot[i * 4 + 1], qy = rot[i * 4 + 2], qz = rot[i * 4 + 3];
  float qn = sqrtf(qw * qw + qx * qx + qy * qy + qz * qz) + 1e-8f;
  float inv = 1.0f / qn;
  qw *= inv; qx *= inv; qy *= inv; qz *= inv;

  float Rq[3][3];
  Rq[0][0] = 1.f - 2.f * (qy * qy + qz * qz); Rq[0][1] = 2.f * (qx * qy - qw * qz); Rq[0][2] = 2.f * (qx * qz + qw * qy);
  Rq[1][0] = 2.f * (qx * qy + qw * qz); Rq[1][1] = 1.f - 2.f * (qx * qx + qz * qz); Rq[1][2] = 2.f * (qy * qz - qw * qx);
  Rq[2][0] = 2.f * (qx * qz - qw * qy); Rq[2][1] = 2.f * (qy * qz + qw * qx); Rq[2][2] = 1.f - 2.f * (qx * qx + qy * qy);

  float s0 = scales[i * 3 + 0], s1 = scales[i * 3 + 1], s2 = scales[i * 3 + 2];
  float M[3][3];
#pragma unroll
  for (int r = 0; r < 3; ++r) { M[r][0] = Rq[r][0] * s0; M[r][1] = Rq[r][1] * s1; M[r][2] = Rq[r][2] * s2; }
  float Sig[3][3];
#pragma unroll
  for (int r = 0; r < 3; ++r)
#pragma unroll
    for (int c = 0; c < 3; ++c)
      Sig[r][c] = M[r][0] * M[c][0] + M[r][1] * M[c][1] + M[r][2] * M[c][2];

  float T1[3][3];
#pragma unroll
  for (int r = 0; r < 3; ++r)
#pragma unroll
    for (int c = 0; c < 3; ++c)
      T1[r][c] = Rw[r][0] * Sig[0][c] + Rw[r][1] * Sig[1][c] + Rw[r][2] * Sig[2][c];
  float Sc[3][3];
#pragma unroll
  for (int r = 0; r < 3; ++r)
#pragma unroll
    for (int c = 0; c < 3; ++c)
      Sc[r][c] = T1[r][0] * Rw[c][0] + T1[r][1] * Rw[c][1] + T1[r][2] * Rw[c][2];

  float J00 = fx / tzs, J02 = -fx * tcx / (tzs * tzs);
  float J11 = fy / tzs, J12 = -fy * tcy / (tzs * tzs);
  float JS00 = J00 * Sc[0][0] + J02 * Sc[2][0];
  float JS01 = J00 * Sc[0][1] + J02 * Sc[2][1];
  float JS02 = J00 * Sc[0][2] + J02 * Sc[2][2];
  float JS11 = J11 * Sc[1][1] + J12 * Sc[2][1];
  float JS12 = J11 * Sc[1][2] + J12 * Sc[2][2];
  float a = JS00 * J00 + JS02 * J02 + 0.3f;
  float b = JS01 * J11 + JS02 * J12;
  float c = JS11 * J11 + JS12 * J12 + 0.3f;

  float invdet = 1.0f / fmaxf(a * c - b * b, 1e-12f);
  float Ac = c * invdet, Bc = -b * invdet, Cc = a * invdet;
  float u = fx * tcx / tzs + cx;
  float v = fy * tcy / tzs + cy;

  float key = valid ? tcz : __builtin_inff();
  keys[i] = key;
  __syncthreads();
  int rank = 0;
  for (int j = 0; j < NG; ++j) {
    float kj = keys[j];
    rank += (kj < key) || (kj == key && j < i);
  }

  float* par = (float*)(ws + OFF_PAR);
  par[0 * NG + rank] = 0.5f * Ac;
  par[1 * NG + rank] = 0.5f * Cc;
  par[2 * NG + rank] = Bc;
  par[3 * NG + rank] = u;
  par[4 * NG + rank] = v;
  par[5 * NG + rank] = valid ? opac[i] : 0.0f;
  par[6 * NG + rank] = valid ? tcz : 0.0f;

  _Float16* ft = (_Float16*)(ws + OFF_FEATT);
  for (int ch = 0; ch < CF; ++ch) ft[ch * NG + rank] = (_Float16)features[i * CF + ch];
}

// ================= prep kernel 2: f16-transposed MLP weights =================
__global__ void prep_weights(const float* __restrict__ w1, const float* __restrict__ w2,
                             const float* __restrict__ w3, char* __restrict__ ws) {
  int idx = blockIdx.x * 256 + threadIdx.x;
  _Float16* w1T = (_Float16*)(ws + OFF_W1T);
  _Float16* w2T = (_Float16*)(ws + OFF_W2T);
  _Float16* w3T = (_Float16*)(ws + OFF_W3T);
  if (idx < 32768) {                       // 128 x 256
    int n = idx >> 8, k = idx & 255;
    w1T[n * 256 + k] = (_Float16)w1[k * 128 + n];
  } else if (idx < 49152) {                // 128 x 128
    int j = idx - 32768; int n = j >> 7, k = j & 127;
    w2T[n * 128 + k] = (_Float16)w2[k * 128 + n];
  } else if (idx < 51200) {                // 16 x 128 (pad)
    int j = idx - 49152; int n = j >> 7, k = j & 127;
    w3T[n * 128 + k] = (n < 12) ? (_Float16)w3[k * 12 + n] : (_Float16)0.0f;
  }
}

// ================= fused render + feature GEMM + MLP =================
__global__ __launch_bounds__(256) void render_fused(const char* __restrict__ ws,
                                                    const float* __restrict__ b1,
                                                    const float* __restrict__ b2,
                                                    const float* __restrict__ b3,
                                                    float* __restrict__ out) {
  extern __shared__ char smem[];
  _Float16* wgtL  = (_Float16*)(smem + L_WGT);
  _Float16* rendL = (_Float16*)(smem + L_REND);
  _Float16* h1L   = (_Float16*)(smem + L_H1);
  _Float16* h2L   = (_Float16*)(smem + L_H2);
  float*    scanL = (float*)(smem + L_SCAN);
  float*    depthL= (float*)(smem + L_DEPTH);
  _Float16* featL = (_Float16*)(smem + L_FEAT);

  const int t   = threadIdx.x;
  const int P0  = blockIdx.x * 32;           // 32 pixels per block, within one image row
  const int row = P0 / W_IMG;
  const int col0= P0 % W_IMG;
  const float* par = (const float*)(ws + OFF_PAR);

  if (t < 32) depthL[t] = 0.0f;
  __syncthreads();

  // ---------------- Phase A: alphas + transmittance scan -> wgt (f16, LDS) ----------------
  const int pix = t & 31;                                    // lane id == pixel
  const int seg = __builtin_amdgcn_readfirstlane(t >> 5);    // wave id, force SGPR -> s_load params
  const int g0  = seg * 32;
  const float px = (float)(col0 + pix) + 0.5f;
  const float py = (float)row + 0.5f;

  float al[32];
  float prod = 1.0f;
#pragma unroll
  for (int k = 0; k < 32; ++k) {
    int g = g0 + k;
    float A2 = par[g], C2 = par[NG + g], Bb = par[2 * NG + g];
    float uu = par[3 * NG + g], vv = par[4 * NG + g], op = par[5 * NG + g];
    float dx = px - uu, dy = py - vv;
    float pw = -(A2 * dx * dx + C2 * dy * dy + Bb * dx * dy);
    pw = fminf(pw, 0.0f);
    float a = fminf(0.99f, op * __expf(pw));
    a = (a >= (1.0f / 255.0f)) ? a : 0.0f;
    al[k] = a;
    prod *= (1.0f - a);
  }
  scanL[pix * 9 + seg] = prod;
  __syncthreads();

  float T = 1.0f;
  for (int s = 0; s < seg; ++s) T *= scanL[pix * 9 + s];
  float dsum = 0.0f;
#pragma unroll
  for (int k = 0; k < 32; ++k) {
    int g = g0 + k;
    float a = al[k];
    float w = T * a;
    T *= (1.0f - a);
    wgtL[pix * FSTR + g] = (_Float16)w;
    dsum += w * par[6 * NG + g];
  }
  atomicAdd(&depthL[pix], dsum);

  // ---------------- stage sorted featsT (f16) into padded LDS rows ----------------
  {
    const uint4* fg = (const uint4*)(ws + OFF_FEATT);   // 32 uint4 per (unpadded) row
    uint4* fl = (uint4*)featL;                          // 33 uint4 per (padded) row
#pragma unroll 4
    for (int i = 0; i < 32; ++i) {
      int j = i * 256 + t;
      int ch = j >> 5, pos = j & 31;
      fl[ch * 33 + pos] = fg[j];
    }
  }
  __syncthreads();

  // depth output (clipped)
  if (t < 32) {
    float d = depthL[t];
    d = fminf(fmaxf(d, 0.0f), FARV);
    out[(size_t)HW * CF + P0 + t] = d;
  }

  // ---------------- Phase B: feature GEMM  rendered = wgt (32x256) x featsT^T (256x256) ----------------
  const int wave  = seg;                  // SGPR wave id
  const int lane  = t & 31;
  const int g     = lane >> 4;            // half-wave group
  const int nlo   = lane & 15;            // N within tile / M row select
  const int mtile = wave >> 2;            // 0..1 : which 16-pixel block
  const int pixrow= mtile * 16 + nlo;     // A-fragment row (pixel)
  const int ct0   = (wave & 3) * 4;       // 4 channel tiles per wave

  const u32* wgtU  = (const u32*)wgtL;    // FSTRD dwords / row
  const u32* featU = (const u32*)featL;   // FSTRD dwords / row

  v8f facc[4] = {};
#pragma unroll
  for (int kk = 0; kk < 8; ++kk) {
    v16h afr = load_afrag(wgtU, FSTRD, pixrow, kk, g);
#pragma unroll
    for (int c = 0; c < 4; ++c) {
      int ch = (ct0 + c) * 16 + nlo;
      v16h bfr = load_bfrag(featU, FSTRD, ch, kk, g);
      facc[c] = __builtin_amdgcn_wmma_f32_16x16x32_f16(false, afr, false, bfr,
                                                       (short)0, facc[c], false, false);
    }
  }
#pragma unroll
  for (int c = 0; c < 4; ++c) {
    int ch = (ct0 + c) * 16 + nlo;
#pragma unroll
    for (int r = 0; r < 8; ++r) {
      int M = r + 8 * g;
      int pr = P0 + mtile * 16 + M;
      float val = facc[c][r];
      __builtin_nontemporal_store(val, &out[(size_t)pr * CF + ch]);  // streamed, never re-read
      rendL[(mtile * 16 + M) * FSTR + ch] = (_Float16)val;           // f16 copy for MLP
    }
  }
  __syncthreads();

  // ---------------- MLP layer 1: (32x256) x (256x128) + b1, ReLU ----------------
  const u32* rendU = (const u32*)rendL;                 // FSTRD dwords / row
  const u32* w1U   = (const u32*)(ws + OFF_W1T);        // 128 dwords / row
  const int cl0 = (wave & 3) * 2;                       // 2 of 8 n-tiles per wave
  v8f hacc[2] = {};
#pragma unroll
  for (int kk = 0; kk < 8; ++kk) {
    v16h afr = load_afrag(rendU, FSTRD, pixrow, kk, g);
#pragma unroll
    for (int c = 0; c < 2; ++c) {
      int n = (cl0 + c) * 16 + nlo;
      v16h bfr = load_bfrag(w1U, 128, n, kk, g);
      hacc[c] = __builtin_amdgcn_wmma_f32_16x16x32_f16(false, afr, false, bfr,
                                                       (short)0, hacc[c], false, false);
    }
  }
#pragma unroll
  for (int c = 0; c < 2; ++c) {
    int n = (cl0 + c) * 16 + nlo;
    float bb = b1[n];
#pragma unroll
    for (int r = 0; r < 8; ++r) {
      int M = r + 8 * g;
      float hv = fmaxf(hacc[c][r] + bb, 0.0f);
      h1L[(mtile * 16 + M) * HSTR + n] = (_Float16)hv;
    }
  }
  __syncthreads();

  // ---------------- MLP layer 2: (32x128) x (128x128) + b2, ReLU ----------------
  const u32* h1U = (const u32*)h1L;                     // HSTRD dwords / row
  const u32* w2U = (const u32*)(ws + OFF_W2T);          // 64 dwords / row
  v8f kacc[2] = {};
#pragma unroll
  for (int kk = 0; kk < 4; ++kk) {
    v16h afr = load_afrag(h1U, HSTRD, pixrow, kk, g);
#pragma unroll
    for (int c = 0; c < 2; ++c) {
      int n = (cl0 + c) * 16 + nlo;
      v16h bfr = load_bfrag(w2U, 64, n, kk, g);
      kacc[c] = __builtin_amdgcn_wmma_f32_16x16x32_f16(false, afr, false, bfr,
                                                       (short)0, kacc[c], false, false);
    }
  }
#pragma unroll
  for (int c = 0; c < 2; ++c) {
    int n = (cl0 + c) * 16 + nlo;
    float bb = b2[n];
#pragma unroll
    for (int r = 0; r < 8; ++r) {
      int M = r + 8 * g;
      float hv = fmaxf(kacc[c][r] + bb, 0.0f);
      h2L[(mtile * 16 + M) * HSTR + n] = (_Float16)hv;
    }
  }
  __syncthreads();

  // ---------------- MLP layer 3: (32x128) x (128x12->16) + b3 ----------------
  if ((wave & 3) == 0) {                                // one wave per 16-pixel tile
    const u32* h2U = (const u32*)h2L;                   // HSTRD dwords / row
    const u32* w3U = (const u32*)(ws + OFF_W3T);        // 64 dwords / row
    v8f sacc = {};
#pragma unroll
    for (int kk = 0; kk < 4; ++kk) {
      v16h afr = load_afrag(h2U, HSTRD, pixrow, kk, g);
      v16h bfr = load_bfrag(w3U, 64, nlo, kk, g);
      sacc = __builtin_amdgcn_wmma_f32_16x16x32_f16(false, afr, false, bfr,
                                                    (short)0, sacc, false, false);
    }
    if (nlo < 12) {
      float bb = b3[nlo];
      float* segOut = out + (size_t)HW * CF + HW;
#pragma unroll
      for (int r = 0; r < 8; ++r) {
        int M = r + 8 * g;
        int pr = P0 + mtile * 16 + M;
        segOut[(size_t)pr * 12 + nlo] = sacc[r] + bb;
      }
    }
  }
}

extern "C" void kernel_launch(void* const* d_in, const int* in_sizes, int n_in,
                              void* d_out, int out_size, void* d_ws, size_t ws_size,
                              hipStream_t stream) {
  const float* means3d  = (const float*)d_in[0];
  const float* features = (const float*)d_in[1];
  const float* opac     = (const float*)d_in[2];
  const float* scales   = (const float*)d_in[3];
  const float* rot      = (const float*)d_in[4];
  const float* camK     = (const float*)d_in[5];
  const float* pose     = (const float*)d_in[6];
  const float* w1       = (const float*)d_in[7];
  const float* b1       = (const float*)d_in[8];
  const float* w2       = (const float*)d_in[9];
  const float* b2       = (const float*)d_in[10];
  const float* w3       = (const float*)d_in[11];
  const float* b3       = (const float*)d_in[12];
  char* ws   = (char*)d_ws;
  float* out = (float*)d_out;

  prep_gauss<<<1, 256, 0, stream>>>(means3d, features, opac, scales, rot, camK, pose, ws);
  prep_weights<<<200, 256, 0, stream>>>(w1, w2, w3, ws);
  render_fused<<<HW / 32, 256, SMEM_BYTES, stream>>>(ws, b1, b2, b3, out);
}
